// LSTM_86595130622350
// MI455X (gfx1250) — matrix operand order repaired and checked by reading, not compiled, not run on hardware
//
#include <hip/hip_runtime.h>
#include <hip/hip_bf16.h>

// ---------------------------------------------------------------------------
// 2-layer LSTM (H=51), B=2048, L=999 on gfx1250 via WMMA.
// 128 workgroups (one 16-row batch tile each) x 4 waves.
// Wave w owns gate N-tiles {w, 4+w, 8+w, 12+w}  == i/f/g/o for j in [16w,16w+16):
// it does 1/4 of the WMMAs and 1/4 of the cell updates, c-state in its VGPRs.
// h staging double-buffered by time parity -> only 2 barriers per step.
// Bias / x*W_ih1 / b_lin folded into padded K columns of the weights.
// ---------------------------------------------------------------------------

#define Hsz   51
#define Bsz   2048
#define Lsz   999

typedef __attribute__((ext_vector_type(16))) _Float16 v16h;
typedef __attribute__((ext_vector_type(8)))  _Float16 v8h;
typedef __attribute__((ext_vector_type(8)))  float    v8f;

// ---------------- ws layout (bytes) ----------------
static constexpr size_t WS_XT    = 0;
static constexpr size_t WS_OUTT  = (size_t)Lsz * Bsz * 4;            // 8,183,808
static constexpr size_t WS_WBLOB = WS_OUTT * 2;                       // 16,367,616

// ---------------- shared-mem layout (halfs) ----------------
// w1: 16384 | w2: 32768 | wl: 1024 | sh1: 2x1024 | sh2: 2x1024
static constexpr int SMEM_BYTES = (16384 + 32768 + 1024 + 2048 + 2048) * 2;  // 108,544

// ---------------------------------------------------------------------------
__global__ void prep_kernel(const float* __restrict__ Wih1,
                            const float* __restrict__ Whh1,
                            const float* __restrict__ bih1,
                            const float* __restrict__ bhh1,
                            const float* __restrict__ Wih2,
                            const float* __restrict__ Whh2,
                            const float* __restrict__ bih2,
                            const float* __restrict__ bhh2,
                            const float* __restrict__ Wlin,
                            const float* __restrict__ blin,
                            _Float16* __restrict__ w1h,
                            _Float16* __restrict__ w2h,
                            _Float16* __restrict__ wlh) {
    int idx = blockIdx.x * blockDim.x + threadIdx.x;
    // layer-2 combined weights [256][128]: k<64 vs h1 (W_ih2), 64..127 vs h2 (W_hh2),
    // k==127 -> fused bias (A column 127 is constant 1.0)
    if (idx < 32768) {
        int n = idx >> 7, k = idx & 127;
        int g = n >> 6, j = n & 63;
        float v = 0.0f;
        if (j < Hsz) {
            int row = g * Hsz + j;
            if (k < Hsz)                          v = Wih2[row * Hsz + k];
            else if (k >= 64 && k < 64 + Hsz)     v = Whh2[row * Hsz + (k - 64)];
            else if (k == 127)                    v = bih2[row] + bhh2[row];
        }
        w2h[idx] = (_Float16)v;
    }
    // layer-1 weights [256][64]: k<51 -> W_hh1, k==62 -> W_ih1 col, k==63 -> fused bias
    if (idx < 16384) {
        int n = idx >> 6, k = idx & 63;
        int g = n >> 6, j = n & 63;
        float v = 0.0f;
        if (j < Hsz) {
            int row = g * Hsz + j;
            if (k < Hsz)        v = Whh1[row * Hsz + k];
            else if (k == 62)   v = Wih1[row];
            else if (k == 63)   v = bih1[row] + bhh1[row];
        }
        w1h[idx] = (_Float16)v;
    }
    // output weights as B tile [16][64]: n==0 real; k==63 -> b_lin (A col 63 is 1.0)
    if (idx < 1024) {
        int n = idx >> 6, k = idx & 63;
        float v = 0.0f;
        if (n == 0) {
            if (k < Hsz)      v = Wlin[k];
            else if (k == 63) v = blin[0];
        }
        wlh[idx] = (_Float16)v;
    }
}

// ---------------------------------------------------------------------------
// Tiled transpose: src[R][C] -> dst[C][R]
__global__ void transpose_kernel(const float* __restrict__ src,
                                 float* __restrict__ dst, int R, int C) {
    __shared__ float tile[32][33];
    int bx = blockIdx.x * 32;   // C
    int by = blockIdx.y * 32;   // R
    int tx = threadIdx.x, ty = threadIdx.y;
#pragma unroll
    for (int k = 0; k < 32; k += 8) {
        int r = by + ty + k, c = bx + tx;
        if (r < R && c < C) tile[ty + k][tx] = src[(size_t)r * C + c];
    }
    __syncthreads();
#pragma unroll
    for (int k = 0; k < 32; k += 8) {
        int c = bx + ty + k, r = by + tx;
        if (c < C && r < R) dst[(size_t)c * R + r] = tile[tx][ty + k];
    }
}

// ---------------------------------------------------------------------------
#if __has_builtin(__builtin_amdgcn_tanhf)
__device__ __forceinline__ float fast_tanh(float x) { return __builtin_amdgcn_tanhf(x); }
__device__ __forceinline__ float fast_sig(float x) {
    return fmaf(0.5f, __builtin_amdgcn_tanhf(0.5f * x), 0.5f);
}
#else
__device__ __forceinline__ float fast_sig(float x) {
    return __builtin_amdgcn_rcpf(1.0f + __expf(-x));
}
__device__ __forceinline__ float fast_tanh(float x) {
    return 2.0f * __builtin_amdgcn_rcpf(1.0f + __expf(-2.0f * x)) - 1.0f;
}
#endif

// A fragment (16-bit 16x32, ISA §7.12.2): lane L -> M = L&15,
// VGPR0..3 hold K = hf*8 .. +7, VGPR4..7 hold K = 16+hf*8 .. +7.
__device__ __forceinline__ v16h loadA(const _Float16* buf, int m, int kt, int hf) {
    int k0 = kt * 32 + hf * 8;
    v8h lo = *(const v8h*)(buf + m * 64 + k0);
    v8h hi = *(const v8h*)(buf + m * 64 + k0 + 16);
    v16h r;
#pragma unroll
    for (int i = 0; i < 8; ++i) { r[i] = lo[i]; r[i + 8] = hi[i]; }
    return r;
}

// B fragment (16-bit 32x16): lane L -> N = L&15, K = hf*8 .. +7, then 16+hf*8 .. +7.
__device__ __forceinline__ v16h loadB(const _Float16* wp, int n, int stride, int kt, int hf) {
    const _Float16* p = wp + n * stride + kt * 32 + hf * 8;
    v8h lo = *(const v8h*)(p);
    v8h hi = *(const v8h*)(p + 16);
    v16h r;
#pragma unroll
    for (int i = 0; i < 8; ++i) { r[i] = lo[i]; r[i + 8] = hi[i]; }
    return r;
}

#define WMMA_F32_F16(A, Bm, C) \
    __builtin_amdgcn_wmma_f32_16x16x32_f16(false, (A), false, (Bm), (short)0, (C), false, false)

// ---------------------------------------------------------------------------
__global__ __launch_bounds__(128)
void lstm_kernel(const float* __restrict__ xT,        // [L][B]
                 const _Float16* __restrict__ gw,     // f16 weight blob
                 float* __restrict__ outT) {          // [L][B]
    extern __shared__ __align__(16) char smem_raw[];
    _Float16* w1  = (_Float16*)smem_raw;       // [256][64]
    _Float16* w2  = w1 + 16384;                // [256][128]
    _Float16* wl  = w2 + 32768;                // [16][64]
    _Float16* sh1 = wl + 1024;                 // h1 staging [2][16][64] (k62=x, k63=1)
    _Float16* sh2 = sh1 + 2048;                // h2 staging [2][16][64] (k63=1)

    const int tid  = threadIdx.x;
    const int lane = tid & 31;
    const int wid  = tid >> 5;                 // wave id 0..3 == owned j-group q
    const int l15  = lane & 15;
    const int hf   = lane >> 4;
    const int base = blockIdx.x * 16;          // batch row base of this block

    // ---- prologue: weights -> LDS, zero staging, ones columns, x_0 ----
    {
        const float4* src = (const float4*)gw;
        float4* dst = (float4*)w1;
        for (int i = tid; i < 6272; i += 128) dst[i] = src[i];      // 100,352 B
        float4 z; z.x = z.y = z.z = z.w = 0.0f;
        float4* hz = (float4*)sh1;                                   // sh1+sh2 contiguous
        for (int i = tid; i < 256; i += 128) hz[i] = z;              // 4,096 B
    }
    if (tid < 16) {
        sh1[tid * 64 + 63]        = (_Float16)1.0f;
        sh1[1024 + tid * 64 + 63] = (_Float16)1.0f;
        sh2[tid * 64 + 63]        = (_Float16)1.0f;
        sh2[1024 + tid * 64 + 63] = (_Float16)1.0f;
        sh1[tid * 64 + 62]        = (_Float16)xT[base + tid];        // x_0 in buffer 0
    }
    __syncthreads();

    // cell state slice owned by this wave: c[r] == c[m = hf*8+r][j = wid*16+l15]
    float c1[8], c2[8];
#pragma unroll
    for (int i = 0; i < 8; ++i) { c1[i] = 0.0f; c2[i] = 0.0f; }

    for (int t = 0; t < Lsz; ++t) {
        const int p = t & 1;
        _Float16* s1r = sh1 + p * 1024;          // h1_prev, x_t, 1
        _Float16* s1w = sh1 + (p ^ 1) * 1024;    // h1_new (and x_{t+1})
        _Float16* s2r = sh2 + p * 1024;          // h2_prev, 1
        _Float16* s2w = sh2 + (p ^ 1) * 1024;    // h2_new

        // prefetch x_{t+1} (wave 0 only; clamped at the end)
        float xnext = 0.0f;
        if (wid == 0) {
            int tn = (t + 1 < Lsz) ? t + 1 : t;
            xnext = xT[(size_t)tn * Bsz + base + l15];
        }

        // ===== Layer 1 (this wave's i/f/g/o tiles): [h1|x|1] @ [Whh1|Wih1|b1]^T =====
        v8f acc[4];
#pragma unroll
        for (int gi = 0; gi < 4; ++gi)
#pragma unroll
            for (int r = 0; r < 8; ++r) acc[gi][r] = 0.0f;
#pragma unroll
        for (int kt = 0; kt < 2; ++kt) {
            v16h a = loadA(s1r, l15, kt, hf);
#pragma unroll
            for (int gi = 0; gi < 4; ++gi) {
                v16h b = loadB(w1, (gi * 4 + wid) * 16 + l15, 64, kt, hf);
                acc[gi] = WMMA_F32_F16(a, b, acc[gi]);
            }
        }
        // cell update (j = wid*16 + l15); skip j==62 store (x slot), j==63 stays 1.0
#pragma unroll
        for (int r = 0; r < 8; ++r) {
            float c = fast_sig(acc[1][r]) * c1[r] + fast_sig(acc[0][r]) * fast_tanh(acc[2][r]);
            c1[r] = c;
            float h = fast_sig(acc[3][r]) * fast_tanh(c);
            _Float16 st = (_Float16)h;
            bool do_store = true;
            if (wid == 3) {
                if (l15 == 15) st = (_Float16)1.0f;      // keep A[.,63] == 1
                if (l15 == 14) do_store = false;         // x slot owned by wave 0
            }
            if (do_store) s1w[(hf * 8 + r) * 64 + wid * 16 + l15] = st;
        }
        __syncthreads();                                  // h1_new visible

        // stage x_{t+1} into the buffer layer-2 reads (column 62 is dead for layer 2)
        if (wid == 0 && hf == 0) s1w[l15 * 64 + 62] = (_Float16)xnext;

        // ===== Layer 2: [h1_new | h2_prev | 1] @ [Wih2|Whh2|b2]^T =====
#pragma unroll
        for (int gi = 0; gi < 4; ++gi)
#pragma unroll
            for (int r = 0; r < 8; ++r) acc[gi][r] = 0.0f;
#pragma unroll
        for (int kt = 0; kt < 4; ++kt) {
            const _Float16* asrc = (kt < 2) ? s1w : s2r;
            v16h a = loadA(asrc, l15, kt & 1, hf);
#pragma unroll
            for (int gi = 0; gi < 4; ++gi) {
                v16h b = loadB(w2, (gi * 4 + wid) * 16 + l15, 128, kt, hf);
                acc[gi] = WMMA_F32_F16(a, b, acc[gi]);
            }
        }
#pragma unroll
        for (int r = 0; r < 8; ++r) {
            float c = fast_sig(acc[1][r]) * c2[r] + fast_sig(acc[0][r]) * fast_tanh(acc[2][r]);
            c2[r] = c;
            float h = fast_sig(acc[3][r]) * fast_tanh(c);
            _Float16 st = (_Float16)h;
            if (wid == 3 && l15 == 15) st = (_Float16)1.0f;  // keep A[.,63] == 1
            s2w[(hf * 8 + r) * 64 + wid * 16 + l15] = st;
        }
        __syncthreads();                                  // h2_new (and x_{t+1}) visible

        // ===== Output head: out = [h2|1] @ [Wlin|b_lin]^T  (round-robin wave) =====
        if (wid == (t & 3)) {
            v8f acco;
#pragma unroll
            for (int r = 0; r < 8; ++r) acco[r] = 0.0f;
#pragma unroll
            for (int kt = 0; kt < 2; ++kt) {
                v16h a = loadA(s2w, l15, kt, hf);
                v16h b = loadB(wl, l15, 64, kt, hf);      // rows n>0 are zero
                acco = WMMA_F32_F16(a, b, acco);
            }
            if (l15 == 0) {                               // lanes 0 & 16: n==0, m = hf*8+r
#pragma unroll
                for (int r = 0; r < 8; ++r)
                    outT[(size_t)t * Bsz + base + hf * 8 + r] = acco[r];
            }
        }
    }
}

// ---------------------------------------------------------------------------
extern "C" void kernel_launch(void* const* d_in, const int* in_sizes, int n_in,
                              void* d_out, int out_size, void* d_ws, size_t ws_size,
                              hipStream_t stream) {
    const float* x    = (const float*)d_in[0];
    const float* Wih1 = (const float*)d_in[1];
    const float* Whh1 = (const float*)d_in[2];
    const float* bih1 = (const float*)d_in[3];
    const float* bhh1 = (const float*)d_in[4];
    const float* Wih2 = (const float*)d_in[5];
    const float* Whh2 = (const float*)d_in[6];
    const float* bih2 = (const float*)d_in[7];
    const float* bhh2 = (const float*)d_in[8];
    const float* Wlin = (const float*)d_in[9];
    const float* blin = (const float*)d_in[10];

    char* ws = (char*)d_ws;
    float*    xT    = (float*)(ws + WS_XT);
    float*    outT  = (float*)(ws + WS_OUTT);
    _Float16* wblob = (_Float16*)(ws + WS_WBLOB);

    // 1) repack weights (f32 -> padded f16 layout, bias/x/b_lin columns fused)
    prep_kernel<<<(32768 + 255) / 256, 256, 0, stream>>>(
        Wih1, Whh1, bih1, bhh1, Wih2, Whh2, bih2, bhh2, Wlin, blin,
        wblob, wblob + 16384, wblob + 49152);

    // 2) x [B][L] -> xT [L][B] (time-major, coalesced per-step reads)
    transpose_kernel<<<dim3((Lsz + 31) / 32, (Bsz + 31) / 32), dim3(32, 8), 0, stream>>>(
        x, xT, Bsz, Lsz);

    // 3) persistent LSTM: 128 blocks x 4 waves; each WGP runs one 16-row chain
    lstm_kernel<<<Bsz / 16, 128, SMEM_BYTES, stream>>>(xT, wblob, outT);

    // 4) outT [L][B] -> out [B][L]
    transpose_kernel<<<dim3((Bsz + 31) / 32, (Lsz + 31) / 32), dim3(32, 8), 0, stream>>>(
        outT, (float*)d_out, Lsz, Bsz);
}
